// CustomAttn_61443802136603
// MI455X (gfx1250) — matrix operand order repaired and checked by reading, not compiled, not run on hardware
//
#include <hip/hip_runtime.h>
#include <hip/hip_bf16.h>

typedef __attribute__((ext_vector_type(8)))  __bf16 bf16x8;
typedef __attribute__((ext_vector_type(16))) __bf16 bf16x16;
typedef __attribute__((ext_vector_type(8)))  float  f32x8;
typedef __attribute__((ext_vector_type(4)))  int    int4v;

constexpr int Bc  = 4;
constexpr int Sc  = 2048;
constexpr int Dc  = 2048;
constexpr int NHc = 16;
constexpr int HDc = 128;
constexpr float QSCALE = 0.08838834764831845f;  // 1/sqrt(128)

// ---- CDNA5 async global->LDS copy (ASYNCcnt path), with sync fallback ----
#if defined(__has_builtin)
#  if __has_builtin(__builtin_amdgcn_global_load_async_to_lds_b128)
#    define HAVE_ASYNC_LDS 1
#  endif
#endif
#ifndef HAVE_ASYNC_LDS
#  define HAVE_ASYNC_LDS 0
#endif

#if HAVE_ASYNC_LDS
typedef __attribute__((address_space(1))) int4v GInt4;
typedef __attribute__((address_space(3))) int4v LInt4;
__device__ __forceinline__ void async_b128(const void* g, void* l) {
  __builtin_amdgcn_global_load_async_to_lds_b128((GInt4*)g, (LInt4*)l, 0, 0);
}
__device__ __forceinline__ void async_wait() {
  asm volatile("s_wait_asynccnt 0x0" ::: "memory");
}
#endif

__device__ __forceinline__ bf16x16 pack16(bf16x8 lo, bf16x8 hi) {
  bf16x16 r;
#pragma unroll
  for (int i = 0; i < 8; ++i) { r[i] = lo[i]; r[i + 8] = hi[i]; }
  return r;
}

// ---------------------------------------------------------------------------
// f32 -> bf16 elementwise convert
// ---------------------------------------------------------------------------
__global__ __launch_bounds__(256) void cvt_f32_bf16(const float* __restrict__ in,
                                                    __bf16* __restrict__ out, int n) {
  int i = blockIdx.x * 256 + threadIdx.x;
  if (i < n) out[i] = (__bf16)in[i];
}

// ---------------------------------------------------------------------------
// C[M,N] (f32) = A[M,K] (bf16) * Bw[N,K]^T (bf16)  via v_wmma_f32_16x16x32_bf16
// block = 256 threads = 8 waves. Block tile 128x64, K-step 128, LDS staged,
// double buffered (async-to-LDS when available). Wave computes a 16x64 tile.
// LDS->VGPR fragment reads are software-pipelined against the WMMA chain.
// ---------------------------------------------------------------------------
__global__ __launch_bounds__(256) void gemm_bf16_wmma(
    const __bf16* __restrict__ A, const __bf16* __restrict__ Bw,
    float* __restrict__ C, int M, int N, int K) {
  __shared__ __align__(16) __bf16 sA[2][128][128];  // 64 KB
  __shared__ __align__(16) __bf16 sB[2][64][128];   // 32 KB

  const int tid  = threadIdx.x;
  const int lane = tid & 31;
  const int wave = tid >> 5;
  const int hf   = lane >> 4;   // 16-lane half
  const int l16  = lane & 15;
  const int tileN  = blockIdx.x * 64;
  const int tileM0 = blockIdx.y * 128;

  // cooperative 16B-chunk copy coords: row = 128 bf16 = 16 chunks
  const int crow = tid >> 4;         // 0..15
  const int ccol = (tid & 15) * 8;   // bf16 offset within row

  f32x8 acc[4] = {};

  auto compute = [&](int cur) {
    bf16x16 afp[2];
    bf16x16 bfp[2][4];
    auto ldfrag = [&](int pb, int kk) {
      const __bf16* arp = &sA[cur][wave * 16 + l16][kk + hf * 8];
      afp[pb] = pack16(*(const bf16x8*)arp, *(const bf16x8*)(arp + 16));
#pragma unroll
      for (int t = 0; t < 4; ++t) {
        const __bf16* brp = &sB[cur][t * 16 + l16][kk + hf * 8];
        bfp[pb][t] = pack16(*(const bf16x8*)brp, *(const bf16x8*)(brp + 16));
      }
    };
    ldfrag(0, 0);
#pragma unroll
    for (int kk = 0; kk < 4; ++kk) {
      if (kk < 3) ldfrag((kk + 1) & 1, (kk + 1) * 32);  // prefetch next frags
#pragma unroll
      for (int t = 0; t < 4; ++t) {
        acc[t] = __builtin_amdgcn_wmma_f32_16x16x32_bf16(
            false, afp[kk & 1], false, bfp[kk & 1][t], (short)0, acc[t], false, false);
      }
    }
  };

#if HAVE_ASYNC_LDS
  auto copy_tile = [&](int st, int kt) {
#pragma unroll
    for (int i = 0; i < 8; ++i) {   // A: 128 rows
      int r = crow + i * 16;
      async_b128(A + (size_t)(tileM0 + r) * K + kt + ccol, &sA[st][r][ccol]);
    }
#pragma unroll
    for (int i = 0; i < 4; ++i) {   // B: 64 rows
      int r = crow + i * 16;
      async_b128(Bw + (size_t)(tileN + r) * K + kt + ccol, &sB[st][r][ccol]);
    }
  };
  copy_tile(0, 0);
  async_wait();
  __syncthreads();
  int cur = 0;
  for (int kt = 0; kt < K; kt += 128) {
    if (kt + 128 < K) copy_tile(cur ^ 1, kt + 128);
    compute(cur);
    async_wait();
    __syncthreads();
    cur ^= 1;
  }
#else
  bf16x8 ra[8], rb[4];
  auto load_regs = [&](int kt) {
#pragma unroll
    for (int i = 0; i < 8; ++i)
      ra[i] = *(const bf16x8*)(A + (size_t)(tileM0 + crow + i * 16) * K + kt + ccol);
#pragma unroll
    for (int i = 0; i < 4; ++i)
      rb[i] = *(const bf16x8*)(Bw + (size_t)(tileN + crow + i * 16) * K + kt + ccol);
  };
  auto store_regs = [&](int st) {
#pragma unroll
    for (int i = 0; i < 8; ++i) *(bf16x8*)(&sA[st][crow + i * 16][ccol]) = ra[i];
#pragma unroll
    for (int i = 0; i < 4; ++i) *(bf16x8*)(&sB[st][crow + i * 16][ccol]) = rb[i];
  };
  load_regs(0);
  store_regs(0);
  __syncthreads();
  int cur = 0;
  for (int kt = 0; kt < K; kt += 128) {
    const bool nxt = (kt + 128) < K;
    if (nxt) load_regs(kt + 128);     // global loads overlap with LDS compute
    compute(cur);
    if (nxt) store_regs(cur ^ 1);
    __syncthreads();
    cur ^= 1;
  }
#endif

  // C layout: lane holds column tileN + t*16 + l16, rows v + 8*hf
#pragma unroll
  for (int t = 0; t < 4; ++t) {
#pragma unroll
    for (int v = 0; v < 8; ++v) {
      C[(size_t)(tileM0 + wave * 16 + v + 8 * hf) * N + tileN + t * 16 + l16] = acc[t][v];
    }
  }
}

// ---------------------------------------------------------------------------
// LayerNorm(q,k) + RoPE + pack:
//   qp[b,h,s,hd] = rope(LN(q)) * 1/sqrt(HD)   (bf16)
//   kp[b,h,s,hd] = rope(LN(k))                (bf16)
//   vt[b,h,hd,s] = v (transposed, bf16) so PV B-operand is contiguous over s
// ---------------------------------------------------------------------------
__global__ __launch_bounds__(256) void ln_rope_pack(
    const float* __restrict__ qkv,
    const float* __restrict__ qg, const float* __restrict__ qb,
    const float* __restrict__ kg, const float* __restrict__ kb,
    __bf16* __restrict__ qp, __bf16* __restrict__ kp, __bf16* __restrict__ vt) {
  __shared__ float sh[Dc];
  __shared__ float rsum[8], rsq[8];
  __shared__ float sMu, sRstd;

  const int tid  = threadIdx.x;
  const int lane = tid & 31;
  const int wave = tid >> 5;
  const int bs   = blockIdx.x;
  const int b    = bs / Sc;
  const int spos = bs % Sc;
  const float* base = qkv + (size_t)bs * (3 * Dc);

  for (int sec = 0; sec < 2; ++sec) {          // 0 = q, 1 = k
    const float* g  = sec ? kg : qg;
    const float* be = sec ? kb : qb;
    __bf16* dst     = sec ? kp : qp;

    float x[8];
    float s1 = 0.f, s2 = 0.f;
#pragma unroll
    for (int j = 0; j < 8; ++j) {
      x[j] = base[sec * Dc + tid * 8 + j];
      s1 += x[j];
      s2 += x[j] * x[j];
    }
#pragma unroll
    for (int off = 16; off > 0; off >>= 1) {
      s1 += __shfl_xor(s1, off);
      s2 += __shfl_xor(s2, off);
    }
    if (lane == 0) { rsum[wave] = s1; rsq[wave] = s2; }
    __syncthreads();
    if (tid == 0) {
      float t1 = 0.f, t2 = 0.f;
#pragma unroll
      for (int w = 0; w < 8; ++w) { t1 += rsum[w]; t2 += rsq[w]; }
      float mu = t1 / (float)Dc;
      sMu = mu;
      sRstd = rsqrtf(t2 / (float)Dc - mu * mu + 1e-5f);
    }
    __syncthreads();
    const float mu = sMu, rs = sRstd;
#pragma unroll
    for (int j = 0; j < 8; ++j) {
      int d = tid * 8 + j;
      sh[d] = (x[j] - mu) * rs * g[d] + be[d];
    }
    __syncthreads();
#pragma unroll
    for (int j = 0; j < 8; ++j) {
      int d  = tid * 8 + j;
      int hd = d & (HDc - 1);
      int i  = hd & 63;
      float inv = __powf(10000.0f, -(float)i * (1.0f / 64.0f));
      float sn, cs;
      __sincosf((float)spos * inv, &sn, &cs);
      float part = sh[d ^ 64];
      float rot  = (hd < 64) ? -part : part;
      float val  = sh[d] * cs + rot * sn;
      if (sec == 0) val *= QSCALE;
      int h = d >> 7;
      dst[(((size_t)(b * NHc + h)) * Sc + spos) * HDc + hd] = (__bf16)val;
    }
    __syncthreads();
  }
  // V: transposed store
#pragma unroll
  for (int j = 0; j < 8; ++j) {
    int d  = tid * 8 + j;
    float v = base[2 * Dc + d];
    int h = d >> 7, hd = d & (HDc - 1);
    vt[(((size_t)(b * NHc + h)) * HDc + hd) * Sc + spos] = (__bf16)v;
  }
}

// ---------------------------------------------------------------------------
// Causal flash attention, one wave = 16 query rows, 32 keys per iteration.
// grid = (B*NH, S/128), block = 256. Q pre-scaled by 1/sqrt(HD).
// No block barriers inside the key loop (waves have different causal trip
// counts); the P-transpose staging buffer is per-wave private LDS.
// K/V fragment loads are hoisted ahead of the WMMA chains for overlap.
// ---------------------------------------------------------------------------
__global__ __launch_bounds__(256) void flash_attn(
    const __bf16* __restrict__ qp, const __bf16* __restrict__ kp,
    const __bf16* __restrict__ vt, __bf16* __restrict__ ao) {
  __shared__ __align__(16) __bf16 plds[8 * 16 * 48];

  const int lane = threadIdx.x & 31;
  const int wave = threadIdx.x >> 5;
  const int hf   = lane >> 4;
  const int l16  = lane & 15;

  const int bh = blockIdx.x;
  const int b  = bh / NHc;
  const int h  = bh % NHc;
  const int qbase = blockIdx.y * 128 + wave * 16;

  const __bf16* Qb = qp + (size_t)bh * Sc * HDc;
  const __bf16* Kb = kp + (size_t)bh * Sc * HDc;
  const __bf16* Vb = vt + (size_t)bh * HDc * Sc;
  __bf16* pl = plds + wave * 16 * 48;

  // Q fragments (A layout), K dim = HD = 128 -> 4 fragments of 32
  bf16x16 qf[4];
  {
    const __bf16* qrow = Qb + (size_t)(qbase + l16) * HDc;
#pragma unroll
    for (int f = 0; f < 4; ++f) {
      int k0 = f * 32 + hf * 8;
      qf[f] = pack16(*(const bf16x8*)(qrow + k0),
                     *(const bf16x8*)(qrow + k0 + 16));
    }
  }

  f32x8 o[8] = {};
  float m[8], l[8];
#pragma unroll
  for (int v = 0; v < 8; ++v) { m[v] = -1e30f; l[v] = 0.f; }

  const int jend = (qbase + 16 + 31) >> 5;
  for (int jb = 0; jb < jend; ++jb) {
    const int kbase = jb * 32;

    // ---- hoisted K fragment loads (one clause), then QK^T WMMA chain ----
    bf16x8 klo[8], khi[8];
#pragma unroll
    for (int hv = 0; hv < 2; ++hv) {
      const __bf16* krow = Kb + (size_t)(kbase + hv * 16 + l16) * HDc + hf * 8;
#pragma unroll
      for (int f = 0; f < 4; ++f) {
        klo[hv * 4 + f] = *(const bf16x8*)(krow + f * 32);
        khi[hv * 4 + f] = *(const bf16x8*)(krow + f * 32 + 16);
      }
    }
    f32x8 sc[2] = {};
#pragma unroll
    for (int hv = 0; hv < 2; ++hv) {
#pragma unroll
      for (int f = 0; f < 4; ++f) {
        sc[hv] = __builtin_amdgcn_wmma_f32_16x16x32_bf16(
            false, qf[f], false, pack16(klo[hv * 4 + f], khi[hv * 4 + f]),
            (short)0, sc[hv], false, false);
      }
    }

    // ---- causal mask + online softmax (rows per-lane, cols across half) ----
    const int key0 = kbase + l16;
    const int key1 = kbase + 16 + l16;
#pragma unroll
    for (int v = 0; v < 8; ++v) {
      const int qrow = qbase + v + 8 * hf;
      if (key0 > qrow) sc[0][v] = -1e30f;
      if (key1 > qrow) sc[1][v] = -1e30f;
      float mx = fmaxf(sc[0][v], sc[1][v]);
#pragma unroll
      for (int off = 1; off < 16; off <<= 1) mx = fmaxf(mx, __shfl_xor(mx, off));
      const float mn = fmaxf(m[v], mx);
      const float alpha = __expf(m[v] - mn);
      const float p0 = __expf(sc[0][v] - mn);
      const float p1 = __expf(sc[1][v] - mn);
      sc[0][v] = p0;
      sc[1][v] = p1;
      float ls = p0 + p1;
#pragma unroll
      for (int off = 1; off < 16; off <<= 1) ls += __shfl_xor(ls, off);
      l[v] = l[v] * alpha + ls;
      m[v] = mn;
#pragma unroll
      for (int t = 0; t < 8; ++t) o[t][v] *= alpha;
    }

    // ---- transpose P (C layout -> A layout) through per-wave LDS ----
#pragma unroll
    for (int v = 0; v < 8; ++v) {
      const int r = v + 8 * hf;
      pl[r * 48 + l16]      = (__bf16)sc[0][v];
      pl[r * 48 + 16 + l16] = (__bf16)sc[1][v];
    }

    // ---- hoisted V fragment loads (global, overlap with DS wait) ----
    bf16x8 vlo[8], vhi[8];
#pragma unroll
    for (int t = 0; t < 8; ++t) {
      const __bf16* vrow = Vb + (size_t)(t * 16 + l16) * Sc + kbase + hf * 8;
      vlo[t] = *(const bf16x8*)(vrow);
      vhi[t] = *(const bf16x8*)(vrow + 16);
    }

    bf16x16 pf = pack16(*(const bf16x8*)(pl + l16 * 48 + hf * 8),
                        *(const bf16x8*)(pl + l16 * 48 + hf * 8 + 16));

    // ---- PV WMMA chain ----
#pragma unroll
    for (int t = 0; t < 8; ++t) {
      o[t] = __builtin_amdgcn_wmma_f32_16x16x32_bf16(
          false, pf, false, pack16(vlo[t], vhi[t]), (short)0, o[t], false, false);
    }
  }

  // ---- epilogue: normalize and store ao[b, qrow, h*HD + d] ----
  float linv[8];
#pragma unroll
  for (int v = 0; v < 8; ++v) linv[v] = 1.0f / l[v];
#pragma unroll
  for (int t = 0; t < 8; ++t) {
#pragma unroll
    for (int v = 0; v < 8; ++v) {
      const int qrow = qbase + v + 8 * hf;
      ao[((size_t)b * Sc + qrow) * Dc + h * HDc + t * 16 + l16] =
          (__bf16)(o[t][v] * linv[v]);
    }
  }
}

// ---------------------------------------------------------------------------
// Host launcher
// ---------------------------------------------------------------------------
extern "C" void kernel_launch(void* const* d_in, const int* in_sizes, int n_in,
                              void* d_out, int out_size, void* d_ws, size_t ws_size,
                              hipStream_t stream) {
  (void)in_sizes; (void)n_in; (void)out_size; (void)ws_size;

  const float* x     = (const float*)d_in[0];
  const float* w_in  = (const float*)d_in[1];
  const float* w_out = (const float*)d_in[2];
  const float* qg    = (const float*)d_in[3];
  const float* qbta  = (const float*)d_in[4];
  const float* kg    = (const float*)d_in[5];
  const float* kbta  = (const float*)d_in[6];
  float* out = (float*)d_out;

  char* ws = (char*)d_ws;
  size_t off = 0;
  auto take = [&](size_t bytes) -> char* {
    char* p = ws + off;
    off += (bytes + 255) & ~(size_t)255;
    return p;
  };

  __bf16* x_bf    = (__bf16*)take((size_t)Bc * Sc * Dc * 2);
  __bf16* win_bf  = (__bf16*)take((size_t)3 * Dc * Dc * 2);
  __bf16* wout_bf = (__bf16*)take((size_t)Dc * Dc * 2);
  float*  qkv     = (float*) take((size_t)Bc * Sc * 3 * Dc * 4);
  __bf16* qp      = (__bf16*)take((size_t)Bc * Sc * Dc * 2);
  __bf16* kp      = (__bf16*)take((size_t)Bc * Sc * Dc * 2);
  __bf16* vt      = (__bf16*)take((size_t)Bc * Sc * Dc * 2);
  __bf16* ao      = (__bf16*)take((size_t)Bc * Sc * Dc * 2);

  int n;
  n = Bc * Sc * Dc;
  cvt_f32_bf16<<<(n + 255) / 256, 256, 0, stream>>>(x, x_bf, n);
  n = 3 * Dc * Dc;
  cvt_f32_bf16<<<(n + 255) / 256, 256, 0, stream>>>(w_in, win_bf, n);
  n = Dc * Dc;
  cvt_f32_bf16<<<(n + 255) / 256, 256, 0, stream>>>(w_out, wout_bf, n);

  {
    dim3 g((3 * Dc) / 64, (Bc * Sc) / 128);
    gemm_bf16_wmma<<<g, 256, 0, stream>>>(x_bf, win_bf, qkv, Bc * Sc, 3 * Dc, Dc);
  }

  ln_rope_pack<<<Bc * Sc, 256, 0, stream>>>(qkv, qg, qbta, kg, kbta, qp, kp, vt);

  flash_attn<<<dim3(Bc * NHc, Sc / 128), 256, 0, stream>>>(qp, kp, vt, ao);

  {
    dim3 g(Dc / 64, (Bc * Sc) / 128);
    gemm_bf16_wmma<<<g, 256, 0, stream>>>(ao, wout_bf, out, Bc * Sc, Dc, Dc);
  }
}
